// CausalSelfAttention_63290638074118
// MI455X (gfx1250) — matrix operand order repaired and checked by reading, not compiled, not run on hardware
//
#include <hip/hip_runtime.h>
#include <math.h>

// ---------------------------------------------------------------------------
// CausalSelfAttention forward for MI455X (gfx1250, wave32, WMMA).
//   1) convert x -> bf16 ; transpose+convert W_attn, W_proj -> bf16 [N][K]
//   2) GEMM1: qkv = x @ W_attn + b (bf16 WMMA, fp32 acc); scatter q/k [B,H,T,hd],
//      v transposed [B,H,hd,T] (q pre-scaled by 1/sqrt(hd))
//   3) flash attention, double-buffered async K/V tiles, WMMA for S and P@V
//   4) GEMM2: out = y @ W_proj + b_proj (fp32 output)
// Global->LDS staging uses gfx1250 GLOBAL_LOAD_ASYNC_TO_LDS_B128 with
// s_wait_asynccnt pipelined double buffering.
// ---------------------------------------------------------------------------

typedef __attribute__((ext_vector_type(16))) __bf16 v16bf;
typedef __attribute__((ext_vector_type(8)))  float  v8f;

struct alignas(16) U4 { unsigned int x, y, z, w; };
union FragB { U4 u[2]; v16bf v; };            // 32B fragment (16 bf16)

typedef __attribute__((address_space(3))) char lds_char;

#define WAIT_ASYNCCNT(N) asm volatile("s_wait_asynccnt " #N ::: "memory")

// One 16B global -> LDS async copy per lane (ASYNCcnt-tracked DMA).
__device__ __forceinline__ void async_copy_b128(const void* gptr, void* lptr) {
  unsigned lds = (unsigned)(uintptr_t)(lds_char*)lptr;       // raw LDS byte addr
  unsigned long long ga = (unsigned long long)(uintptr_t)gptr;
  asm volatile("global_load_async_to_lds_b128 %0, %1, off"
               :: "v"(lds), "v"(ga) : "memory");
}

__device__ __forceinline__ unsigned short f2bf(float f) {
  unsigned int u = __float_as_uint(f);
  unsigned int r = (u + 0x7FFFu + ((u >> 16) & 1u)) >> 16;
  return (unsigned short)r;
}

__device__ __forceinline__ v8f v8f_zero() {
  v8f z;
#pragma unroll
  for (int i = 0; i < 8; ++i) z[i] = 0.0f;
  return z;
}

// ------------------------- conversion kernels ------------------------------

__global__ __launch_bounds__(256) void f32_to_bf16_k(
    const float* __restrict__ in, unsigned short* __restrict__ out, int n4) {
  int i = blockIdx.x * blockDim.x + threadIdx.x;
  if (i >= n4) return;
  float4 f = ((const float4*)in)[i];
  unsigned int lo = (unsigned int)f2bf(f.x) | ((unsigned int)f2bf(f.y) << 16);
  unsigned int hi = (unsigned int)f2bf(f.z) | ((unsigned int)f2bf(f.w) << 16);
  ((uint2*)out)[i] = make_uint2(lo, hi);
}

// W [R][Cc] fp32 row-major  ->  Wt [Cc][R] bf16 row-major (K-contiguous)
__global__ __launch_bounds__(1024) void transpose_bf16_k(
    const float* __restrict__ W, unsigned short* __restrict__ Wt, int R, int Cc) {
  __shared__ float tile[32][33];
  int j = blockIdx.x * 32 + threadIdx.x;   // col of W
  int i = blockIdx.y * 32 + threadIdx.y;   // row of W
  tile[threadIdx.y][threadIdx.x] = W[(size_t)i * Cc + j];
  __syncthreads();
  int jo = blockIdx.x * 32 + threadIdx.y;  // row of Wt
  int io = blockIdx.y * 32 + threadIdx.x;  // col of Wt
  Wt[(size_t)jo * R + io] = f2bf(tile[threadIdx.x][threadIdx.y]);
}

// ------------------------------- GEMM --------------------------------------
// A [M][K] bf16 row-major, Bt [N][K] bf16 row-major (transposed weights).
// Block tile 64(M) x 128(N), 8 waves (4x2), wave tile 16x64, K-step 32.
// Double-buffered async global->LDS staging.
// MODE 0: qkv epilogue (bias + scatter; q scaled 0.125; v transposed)
// MODE 1: fp32 output = acc + bias

template <int MODE>
__global__ __launch_bounds__(256) void gemm_bf16_k(
    const unsigned short* __restrict__ A,
    const unsigned short* __restrict__ Bt,
    const float* __restrict__ bias,
    unsigned short* __restrict__ outq,
    unsigned short* __restrict__ outk,
    unsigned short* __restrict__ outv,
    float* __restrict__ Cout,
    int M, int N, int K) {
  __shared__ unsigned short As[2][64 * 32];
  __shared__ unsigned short Bs[2][128 * 32];

  const int tid  = threadIdx.x;
  const int lane = tid & 31;
  const int wid  = tid >> 5;
  const int half = lane >> 4;
  const int l15  = lane & 15;
  const int wm   = wid & 3;       // 4 wave-rows of 16
  const int wn   = wid >> 2;      // 2 wave-cols of 64
  const int m0   = blockIdx.x * 64;
  const int n0   = blockIdx.y * 128;

  auto issue_tiles = [&](int buf, int k0) {
    int r = tid >> 2, ca = (tid & 3) * 8;          // A tile 64x32: 16B/thread
    async_copy_b128(&A[(size_t)(m0 + r) * K + k0 + ca], &As[buf][r * 32 + ca]);
    int n = tid >> 1, cb = (tid & 1) * 16;         // B tile 128x32: 32B/thread
    async_copy_b128(&Bt[(size_t)(n0 + n) * K + k0 + cb],     &Bs[buf][n * 32 + cb]);
    async_copy_b128(&Bt[(size_t)(n0 + n) * K + k0 + cb + 8], &Bs[buf][n * 32 + cb + 8]);
  };

  v8f acc[4];
#pragma unroll
  for (int nt = 0; nt < 4; ++nt) acc[nt] = v8f_zero();

  const int niter = K >> 5;
  issue_tiles(0, 0);
  for (int i = 0; i < niter; ++i) {
    const int cur = i & 1;
    __syncthreads();                       // prev compute done: safe to overwrite
    if (i + 1 < niter) {
      issue_tiles(cur ^ 1, (i + 1) * 32);  // prefetch next tile
      WAIT_ASYNCCNT(3);                    // oldest 3 (current tile) landed
    } else {
      WAIT_ASYNCCNT(0);
    }
    __syncthreads();                       // current tile visible to all waves

    // A fragment: lane = row, elems = K(16*half + 0..15), K-contiguous
    FragB a;
    int am = wm * 16 + l15;
    a.u[0] = *(const U4*)&As[cur][am * 32 + 16 * half];
    a.u[1] = *(const U4*)&As[cur][am * 32 + 16 * half + 8];

#pragma unroll
    for (int nt = 0; nt < 4; ++nt) {
      // B fragment: lane = column, elems = K(16*half + 0..15), K-contiguous
      FragB b;
      int bn = wn * 64 + nt * 16 + l15;
      b.u[0] = *(const U4*)&Bs[cur][bn * 32 + 16 * half];
      b.u[1] = *(const U4*)&Bs[cur][bn * 32 + 16 * half + 8];
      acc[nt] = __builtin_amdgcn_wmma_f32_16x16x32_bf16(
          false, a.v, false, b.v, (short)0, acc[nt], false, false);
    }
  }

#pragma unroll
  for (int nt = 0; nt < 4; ++nt) {
    int nn = n0 + wn * 64 + nt * 16 + l15;
    float bsv = bias[nn];
#pragma unroll
    for (int r = 0; r < 8; ++r) {
      int m = m0 + wm * 16 + r + 8 * half;    // C layout: row = r + 8*(lane>=16)
      float val = acc[nt][r] + bsv;
      if constexpr (MODE == 0) {
        int which = nn >> 10;                 // 0=q 1=k 2=v
        int c = nn & 1023;
        int h = c >> 6, d = c & 63;
        int bb = m >> 11, tt = m & 2047;      // T = 2048
        if (which == 0) {
          outq[(((size_t)(bb * 16 + h)) * 2048 + tt) * 64 + d] = f2bf(val * 0.125f);
        } else if (which == 1) {
          outk[(((size_t)(bb * 16 + h)) * 2048 + tt) * 64 + d] = f2bf(val);
        } else {                              // v stored transposed: [B,H,hd,T]
          outv[(((size_t)(bb * 16 + h)) * 64 + d) * 2048 + tt] = f2bf(val);
        }
      } else {
        Cout[(size_t)m * N + nn] = val;
      }
    }
  }
}

// --------------------------- flash attention -------------------------------
// grid (T/128, H, B), block 256. Wave w handles q rows [q0+16w, q0+16w+16).
// q/k: bf16 [B,H,T,64]; v: bf16 [B,H,64,T] (transposed); y: bf16 [B*T, C].
// K/V tiles double-buffered via async global->LDS copies.

__global__ __launch_bounds__(256) void attn_k(
    const unsigned short* __restrict__ q,
    const unsigned short* __restrict__ k,
    const unsigned short* __restrict__ v,
    unsigned short* __restrict__ y) {
  __shared__ unsigned short Ks[2][64 * 64];    // [kt][d]   (d contiguous)
  __shared__ unsigned short Vt[2][64 * 64];    // [d][kt]   (kt contiguous)
  __shared__ unsigned short Ps[8 * 16 * 64];   // per-wave P tile [16][64]

  const int tid  = threadIdx.x;
  const int lane = tid & 31;
  const int wid  = tid >> 5;
  const int half = lane >> 4;
  const int l15  = lane & 15;

  const int q0  = blockIdx.x * 128;
  const int h   = blockIdx.y;
  const int b   = blockIdx.z;
  const int q0w = q0 + wid * 16;
  const size_t bh = (size_t)(b * 16 + h) * 2048 * 64;
  const unsigned short* qg = q + bh;
  const unsigned short* kg = k + bh;
  const unsigned short* vg = v + bh;           // [64][2048]

  auto issue_tiles = [&](int buf, int kt0) {
    int r = tid >> 2, c = (tid & 3) * 16;      // 64 rows x 64 cols, 32B/thread
    async_copy_b128(&kg[(size_t)(kt0 + r) * 64 + c],     &Ks[buf][r * 64 + c]);
    async_copy_b128(&kg[(size_t)(kt0 + r) * 64 + c + 8], &Ks[buf][r * 64 + c + 8]);
    async_copy_b128(&vg[(size_t)r * 2048 + kt0 + c],     &Vt[buf][r * 64 + c]);
    async_copy_b128(&vg[(size_t)r * 2048 + kt0 + c + 8], &Vt[buf][r * 64 + c + 8]);
  };

  // q fragments for the two K=32 steps over hd=64 (kept in registers)
  FragB aq[2];
  {
    const unsigned short* qr = qg + (size_t)(q0w + l15) * 64;
    aq[0].u[0] = *(const U4*)&qr[16 * half];
    aq[0].u[1] = *(const U4*)&qr[16 * half + 8];
    aq[1].u[0] = *(const U4*)&qr[32 + 16 * half];
    aq[1].u[1] = *(const U4*)&qr[32 + 16 * half + 8];
  }

  v8f o[4];
#pragma unroll
  for (int nt = 0; nt < 4; ++nt) o[nt] = v8f_zero();
  float mrow[8], lrow[8];
#pragma unroll
  for (int r = 0; r < 8; ++r) { mrow[r] = -1e30f; lrow[r] = 0.0f; }

  const int nblk = (q0 + 128) / 64;            // causal horizon for this block
  issue_tiles(0, 0);
  for (int blk = 0; blk < nblk; ++blk) {
    const int kt0 = blk * 64;
    const int cur = blk & 1;
    __syncthreads();                           // prev compute done
    if (blk + 1 < nblk) {
      issue_tiles(cur ^ 1, kt0 + 64);          // prefetch next K/V tile
      WAIT_ASYNCCNT(4);                        // oldest 4 (current tile) landed
    } else {
      WAIT_ASYNCCNT(0);
    }
    __syncthreads();

    if (kt0 > q0w + 15) continue;              // fully masked for this wave

    // ---- S = q @ k^T  (16 x 64), 8 WMMAs ----
    v8f s[4];
#pragma unroll
    for (int nt = 0; nt < 4; ++nt) {
      s[nt] = v8f_zero();
#pragma unroll
      for (int ks = 0; ks < 2; ++ks) {
        FragB bk;                              // column = kt row, K = d slice
        int kr = nt * 16 + l15;
        bk.u[0] = *(const U4*)&Ks[cur][kr * 64 + 32 * ks + 16 * half];
        bk.u[1] = *(const U4*)&Ks[cur][kr * 64 + 32 * ks + 16 * half + 8];
        s[nt] = __builtin_amdgcn_wmma_f32_16x16x32_bf16(
            false, aq[ks].v, false, bk.v, (short)0, s[nt], false, false);
      }
    }

    // ---- causal mask + online softmax (16-lane row groups) ----
    unsigned short* Pw = &Ps[wid * 1024];
#pragma unroll
    for (int r = 0; r < 8; ++r) {
      int qi = q0w + r + 8 * half;
      float sv[4];
      float mloc = -1e30f;
#pragma unroll
      for (int nt = 0; nt < 4; ++nt) {
        int kj = kt0 + nt * 16 + l15;
        float x = s[nt][r];
        if (kj > qi) x = -1e30f;
        sv[nt] = x;
        mloc = fmaxf(mloc, x);
      }
#pragma unroll
      for (int xm = 1; xm < 16; xm <<= 1)
        mloc = fmaxf(mloc, __shfl_xor(mloc, xm, 32));
      float mnew  = fmaxf(mrow[r], mloc);
      float alpha = __expf(mrow[r] - mnew);
      float lsum  = 0.0f;
#pragma unroll
      for (int nt = 0; nt < 4; ++nt) {
        float p = __expf(sv[nt] - mnew);
        lsum += p;
        Pw[(r + 8 * half) * 64 + nt * 16 + l15] = f2bf(p);
      }
#pragma unroll
      for (int xm = 1; xm < 16; xm <<= 1) lsum += __shfl_xor(lsum, xm, 32);
      lrow[r] = lrow[r] * alpha + lsum;
      mrow[r] = mnew;
#pragma unroll
      for (int nt = 0; nt < 4; ++nt) o[nt][r] *= alpha;
    }

    // ---- O += P @ V  (16 x 64), 8 WMMAs ----
#pragma unroll
    for (int ks = 0; ks < 2; ++ks) {
      FragB ap;                                // A layout: lane = row
      ap.u[0] = *(const U4*)&Pw[l15 * 64 + 32 * ks + 16 * half];
      ap.u[1] = *(const U4*)&Pw[l15 * 64 + 32 * ks + 16 * half + 8];
#pragma unroll
      for (int nt = 0; nt < 4; ++nt) {
        FragB bv;                              // column = d, K = kt contiguous
        int dn = nt * 16 + l15;
        bv.u[0] = *(const U4*)&Vt[cur][dn * 64 + 32 * ks + 16 * half];
        bv.u[1] = *(const U4*)&Vt[cur][dn * 64 + 32 * ks + 16 * half + 8];
        o[nt] = __builtin_amdgcn_wmma_f32_16x16x32_bf16(
            false, ap.v, false, bv.v, (short)0, o[nt], false, false);
      }
    }
  }

  // ---- normalize and write y bf16 [B*T][C] ----
#pragma unroll
  for (int nt = 0; nt < 4; ++nt) {
#pragma unroll
    for (int r = 0; r < 8; ++r) {
      int qi = q0w + r + 8 * half;
      float val = o[nt][r] / lrow[r];
      y[((size_t)(b * 2048 + qi)) * 1024 + h * 64 + nt * 16 + l15] = f2bf(val);
    }
  }
}

// ------------------------------- launcher ----------------------------------

extern "C" void kernel_launch(void* const* d_in, const int* in_sizes, int n_in,
                              void* d_out, int out_size, void* d_ws, size_t ws_size,
                              hipStream_t stream) {
  (void)in_sizes; (void)n_in; (void)out_size; (void)ws_size;
  const float* x      = (const float*)d_in[0];   // [2,2048,1024]
  const float* W_attn = (const float*)d_in[1];   // [1024,3072]
  const float* b_attn = (const float*)d_in[2];   // [3072]
  const float* W_proj = (const float*)d_in[3];   // [1024,1024]
  const float* b_proj = (const float*)d_in[4];   // [1024]
  float* out = (float*)d_out;                    // [2,2048,1024]

  char* ws = (char*)d_ws;
  const size_t SZ8 = (size_t)4096 * 1024 * 2;    // 8 MB
  unsigned short* xb  = (unsigned short*)(ws);                         // x bf16
  unsigned short* WAt = (unsigned short*)(ws + SZ8);                   // [3072][1024]
  unsigned short* WPt = (unsigned short*)(ws + SZ8 + 6291456);         // [1024][1024]
  unsigned short* qb  = (unsigned short*)(ws + SZ8 + 6291456 + 2097152);
  unsigned short* kb  = qb + (size_t)4096 * 1024;
  unsigned short* vb  = kb + (size_t)4096 * 1024;                      // [B,H,64,T]
  unsigned short* yb  = vb + (size_t)4096 * 1024;                      // total 48 MB

  // 1) precision staging
  f32_to_bf16_k<<<4096, 256, 0, stream>>>(x, xb, 1048576);
  transpose_bf16_k<<<dim3(96, 32), dim3(32, 32), 0, stream>>>(W_attn, WAt, 1024, 3072);
  transpose_bf16_k<<<dim3(32, 32), dim3(32, 32), 0, stream>>>(W_proj, WPt, 1024, 1024);

  // 2) qkv = x @ W_attn + b_attn  -> scatter q/k [B,H,T,hd], v [B,H,hd,T]
  gemm_bf16_k<0><<<dim3(64, 24), 256, 0, stream>>>(
      xb, WAt, b_attn, qb, kb, vb, nullptr, 4096, 3072, 1024);

  // 3) flash attention -> y bf16 [B*T][C]
  attn_k<<<dim3(16, 16, 2), 256, 0, stream>>>(qb, kb, vb, yb);

  // 4) out = y @ W_proj + b_proj  (fp32)
  gemm_bf16_k<1><<<dim3(64, 8), 256, 0, stream>>>(
      yb, WPt, b_proj, nullptr, nullptr, nullptr, out, 4096, 1024, 1024);
}
